// AdaptiveTokenSelector_86406152061633
// MI455X (gfx1250) — matrix operand order repaired and checked by reading, not compiled, not run on hardware
//
#include <hip/hip_runtime.h>
#include <cstdint>

#define NB   4
#define NS   4096
#define ND   1024
#define MAXK 512

typedef __attribute__((ext_vector_type(16))) _Float16     v16h;
typedef __attribute__((ext_vector_type(8)))  float        v8f;
typedef __attribute__((ext_vector_type(4)))  unsigned int v4u;
typedef __attribute__((ext_vector_type(8)))  int          v8i;
typedef __attribute__((ext_vector_type(4)))  int          v4i;

// ---------------------------------------------------------------------------
// Gate kernel: k_per_query[row] = int(256 + 256*sigmoid(dot(Q[row],W)+b))
// GEMV on v_wmma_f32_16x16x32_f16: one wave handles 16 rows; W occupies
// column N=0 of the B fragment. All loads unconditional (float4), the
// column mask is applied with v_cndmask, not exec branches.
// ---------------------------------------------------------------------------
__global__ void __launch_bounds__(256)
gate_kernel(const float* __restrict__ Q, const float* __restrict__ W,
            const float* __restrict__ b, int* __restrict__ kout) {
  const int tid  = threadIdx.x;
  const int wv   = tid >> 5;          // wave within block (8 waves)
  const int lane = tid & 31;
  const int half = lane >> 4;         // 0: lanes 0-15, 1: lanes 16-31
  const int m    = lane & 15;         // row within 16-row tile
  const int rowbase = blockIdx.x * 128 + wv * 16;
  const float* qrow = Q + (size_t)(rowbase + m) * ND;
  const float bias = b[0];
  const bool col0 = (m == 0);

  v8f c = {};
#pragma unroll 2
  for (int kk = 0; kk < ND; kk += 32) {
    // A fragment: 16-bit A 16x32 layout -> lane holds K runs
    // [half*8, half*8+8) and [16+half*8, 24+half*8): two 8-float runs.
    const float4* qa = (const float4*)(qrow + kk + half * 8);
    const float4* qb = (const float4*)(qrow + kk + 16 + half * 8);
    float4 a0 = qa[0], a1 = qa[1];
    float4 a2 = qb[0], a3 = qb[1];
    // B fragment column 0: same K runs as A: W[kk+half*8..+8) and W[kk+16+half*8..+8)
    const float4* wp = (const float4*)(W + kk + half * 8);
    const float4* wq = (const float4*)(W + kk + 16 + half * 8);
    float4 w0 = wp[0], w1 = wp[1], w2 = wq[0], w3 = wq[1];

    float af[16] = { a0.x,a0.y,a0.z,a0.w, a1.x,a1.y,a1.z,a1.w,
                     a2.x,a2.y,a2.z,a2.w, a3.x,a3.y,a3.z,a3.w };
    float wf[16] = { w0.x,w0.y,w0.z,w0.w, w1.x,w1.y,w1.z,w1.w,
                     w2.x,w2.y,w2.z,w2.w, w3.x,w3.y,w3.z,w3.w };
    v16h a, bb;
#pragma unroll
    for (int h = 0; h < 16; ++h) {
      a[h]  = (_Float16)af[h];
      bb[h] = (_Float16)(col0 ? wf[h] : 0.0f);
    }
    c = __builtin_amdgcn_wmma_f32_16x16x32_f16(false, a, false, bb,
                                               (short)0, c, false, false);
  }
  // D[m][0]: lane0 holds M=0..7 in c[0..7], lane16 holds M=8..15
  if (m == 0) {
#pragma unroll
    for (int r = 0; r < 8; ++r) {
      int row  = rowbase + half * 8 + r;
      float x  = c[r] + bias;
      float im = 1.0f / (1.0f + expf(-x));
      kout[row] = (int)(256.0f + 256.0f * im);   // BASE_K + (MAX_K-BASE_K)*imp
    }
  }
}

// ---------------------------------------------------------------------------
// Top-k kernel: one block per score row. TDM DMA row -> LDS (non-temporal),
// monotonic key transform, 4-pass MSB radix-select for the exact
// 512th-largest threshold (with tie count), compact, bitonic sort, emit.
// ---------------------------------------------------------------------------
__global__ void __launch_bounds__(256)
topk_kernel(const float* __restrict__ scores, float* __restrict__ out) {
  __shared__ unsigned s_keys[NS];
  __shared__ int      s_hist[256];
  __shared__ int      s_scan[256];
  __shared__ unsigned s_sel[MAXK];
  __shared__ int      s_digit, s_newR, s_cntG, s_cntE;

  const int tid = threadIdx.x;
  const int row = blockIdx.x;
  const float* rowp = scores + (size_t)row * NS;

  // ---- Stage row into LDS via Tensor Data Mover (wave 0 issues) ----
  if (tid < 32) {
    // flat LDS pointer: low 32 bits are the LDS byte offset (aperture in high bits)
    unsigned lds_off = (unsigned)(uintptr_t)s_keys;
    unsigned long long ga = (unsigned long long)(uintptr_t)rowp;
    // D# group0: count=1 | lds_addr | global_addr[31:0] | ga[56:32] + type=2
    v4u g0 = { 1u, lds_off, (unsigned)ga, (unsigned)(ga >> 32) | 0x80000000u };
    // D# group1: data_size=4B (2<<16); tensor_dim0=4096 (bits79:48);
    // tensor_dim1=1 (bits111:80); tile_dim0=4096 (bits127:112); tile_dim1=1;
    // tensor_dim0_stride=4096
    v8i g1 = { 0x20000, 0x10000000, 0x10000, 0x10000000, 1, NS, 0, 0 };
    v4i gz4 = { 0, 0, 0, 0 };
    v8i gz8 = { 0, 0, 0, 0, 0, 0, 0, 0 };
    // cpol = 1 -> TH=NT: scores (256MB) streamed once, don't pollute 192MB L2
    __builtin_amdgcn_tensor_load_to_lds(g0, g1, gz4, gz4, gz8, 1);
    __builtin_amdgcn_s_wait_tensorcnt(0);
  }
  if (tid == 0) { s_cntG = 0; s_cntE = 0; }
  __syncthreads();

  // ---- monotonic key transform (order-preserving float -> uint) ----
  for (int n = tid; n < NS; n += 256) {
    unsigned u = s_keys[n];
    unsigned msk = (unsigned)((int)u >> 31) | 0x80000000u;
    s_keys[n] = u ^ msk;
  }

  // ---- 4-pass MSB-first radix select of the top-512 boundary ----
  unsigned prefix = 0;
  int R = MAXK;                      // survivors still needed
  for (int pass = 3; pass >= 0; --pass) {
    const int shift = pass * 8;
    s_hist[tid] = 0;
    __syncthreads();
    for (int n = tid; n < NS; n += 256) {
      unsigned key = s_keys[n];
      bool match = (pass == 3) || ((key >> (shift + 8)) == prefix);
      if (match) atomicAdd(&s_hist[(key >> shift) & 0xFF], 1);
    }
    __syncthreads();
    // suffix sums: s_scan[t] = #elems with digit >= t
    s_scan[tid] = s_hist[tid];
    __syncthreads();
    for (int off = 1; off < 256; off <<= 1) {
      int add = (tid + off < 256) ? s_scan[tid + off] : 0;
      __syncthreads();
      s_scan[tid] += add;
      __syncthreads();
    }
    int above = (tid < 255) ? s_scan[tid + 1] : 0;
    if (s_scan[tid] >= R && above < R) { s_digit = tid; s_newR = R - above; }
    __syncthreads();
    prefix = (prefix << 8) | (unsigned)s_digit;
    R = s_newR;
    __syncthreads();
  }
  const unsigned T = prefix;         // exact 512th-largest key
  const int G = MAXK - R;            // count of keys strictly > T

  // ---- compact: all keys > T, plus exactly R copies of T ----
  for (int n = tid; n < NS; n += 256) {
    unsigned key = s_keys[n];
    if (key > T) {
      int p = atomicAdd(&s_cntG, 1);
      s_sel[p] = key;
    } else if (key == T) {
      int p = atomicAdd(&s_cntE, 1);
      if (p < R) s_sel[G + p] = key;
    }
  }

  // ---- bitonic sort 512 keys descending (1 CE per thread per stage) ----
  for (int k2 = 2; k2 <= MAXK; k2 <<= 1) {
    for (int j = k2 >> 1; j > 0; j >>= 1) {
      __syncthreads();
      int i = 2 * tid - (tid & (j - 1));
      int p = i + j;
      bool desc = ((i & k2) == 0);
      unsigned a = s_sel[i], bsl = s_sel[p];
      if ((a < bsl) == desc) { s_sel[i] = bsl; s_sel[p] = a; }
    }
  }
  __syncthreads();

  // ---- decode keys back to floats, write sorted top-512 ----
  float* orow = out + (size_t)row * MAXK;
  for (int n = tid; n < MAXK; n += 256) {
    unsigned key = s_sel[n];
    unsigned u = key ^ ((key >> 31) ? 0x80000000u : 0xFFFFFFFFu);
    orow[n] = __uint_as_float(u);
  }
}

// ---------------------------------------------------------------------------
extern "C" void kernel_launch(void* const* d_in, const int* in_sizes, int n_in,
                              void* d_out, int out_size, void* d_ws, size_t ws_size,
                              hipStream_t stream) {
  (void)in_sizes; (void)n_in; (void)out_size; (void)d_ws; (void)ws_size;
  const float* Q      = (const float*)d_in[0];
  const float* scores = (const float*)d_in[1];
  const float* W      = (const float*)d_in[2];
  const float* b      = (const float*)d_in[3];

  float* out_vals = (float*)d_out;                               // [4,4096,512] f32
  int*   out_k    = (int*)(out_vals + (size_t)NB * NS * MAXK);   // [4,4096] i32

  gate_kernel<<<dim3((NB * NS) / 128), 256, 0, stream>>>(Q, W, b, out_k);
  topk_kernel<<<dim3(NB * NS), 256, 0, stream>>>(scores, out_vals);
}
